// selfMarginLoss_25881472926361
// MI455X (gfx1250) — compile-verified, matrix operand order checked by reading, not run on hardware
//
#include <hip/hip_runtime.h>

typedef __attribute__((ext_vector_type(16))) _Float16 v16h;
typedef __attribute__((ext_vector_type(8)))  float    v8f;

#define MARGIN 0.1f

// One wave32 per row-batch; each wave processes bPerWave rows of 64 scores.
// Per row: 64x64 masked-relu pair matrix built in WMMA A-layout (f16),
// summed via v_wmma_f32_16x16x32_f16 against an all-ones B matrix.
__global__ void selfMarginLoss_wmma_kernel(const float* __restrict__ scores,
                                           const long long* __restrict__ werRank,
                                           float* __restrict__ partials,
                                           int B, int bPerWave)
{
    const int lane        = threadIdx.x & 31;
    const int waveInBlock = threadIdx.x >> 5;
    const int wavesPerBlk = blockDim.x >> 5;
    const int wid         = blockIdx.x * wavesPerBlk + waveInBlock;

    const int half = lane >> 4;   // 0: lanes 0-15, 1: lanes 16-31
    const int l16  = lane & 15;   // A-matrix row within a 16-row block

    // B matrix = all ones (32x16 f16): D[m][n] += sum_k A[m][k]
    v16h ones;
#pragma unroll
    for (int t = 0; t < 16; ++t) ones[t] = (_Float16)1.0f;

    v8f acc = {};  // f32 16x16 accumulator, chained through C across all rows

    for (int bb = 0; bb < bPerWave; ++bb) {
        const int b = wid * bPerWave + bb;
        const long long base = (long long)b * 64;

        // Gather ranked values: lane holds ranked[lane] and ranked[lane+32]
        const int idx0 = (int)werRank[base + lane];
        const int idx1 = (int)werRank[base + lane + 32];
        const float r0 = scores[base + idx0];
        const float r1 = scores[base + idx1];

        // 64x64 pair matrix = 4 row-blocks (i) x 2 col-blocks (j) of 16x32 A-tiles
#pragma unroll
        for (int bi = 0; bi < 4; ++bi) {
            const int i = bi * 16 + l16;
            // ranked[i]: i in [0,64); bank r0 for i<32, r1 for i>=32
            const float ri = __shfl(bi < 2 ? r0 : r1, i & 31, 32);
#pragma unroll
            for (int bj = 0; bj < 2; ++bj) {
                const float rbank = (bj == 0) ? r0 : r1;
                v16h a;
                // 16-bit A 16x32 layout (ISA 7.12.2):
                //   lanes 0-15:  VGPR p<4 -> K=2p,2p+1 ; p>=4 -> K=16+2(p-4)..
                //   lanes 16-31: same + 8
#pragma unroll
                for (int p = 0; p < 8; ++p) {
                    const int kbase = ((p < 4) ? (2 * p) : (16 + 2 * (p - 4))) + 8 * half;
#pragma unroll
                    for (int h = 0; h < 2; ++h) {
                        const int j  = bj * 32 + kbase + h;      // column index in [0,64)
                        const float rj = __shfl(rbank, j & 31, 32); // ds_bpermute broadcast
                        float v = fmaxf(rj - ri + MARGIN, 0.0f);
                        v = (j > i) ? v : 0.0f;                  // strict upper triangle
                        a[2 * p + h] = (_Float16)v;
                    }
                }
                // D = A x Ones + C  (row sums accumulate into acc)
                acc = __builtin_amdgcn_wmma_f32_16x16x32_f16(
                    /*neg_a=*/false, a, /*neg_b=*/false, ones,
                    /*c_mod=*/(short)0, acc, /*reuse_a=*/false, /*reuse_b=*/false);
            }
        }
    }

    // acc VGPR c at this lane holds D[c + 8*half][l16]; every column of D is
    // bitwise identical (B == ones), so column 0 sum = lane0 + lane16 partials.
    float s = acc[0] + acc[1] + acc[2] + acc[3] + acc[4] + acc[5] + acc[6] + acc[7];
    const float p0  = __shfl(s, 0, 32);
    const float p16 = __shfl(s, 16, 32);
    if (lane == 0) partials[wid] = p0 + p16;
}

// Deterministic single-block tree reduction of the wave partials.
__global__ void reduce_partials_kernel(const float* __restrict__ partials,
                                       float* __restrict__ out, int n)
{
    __shared__ float sm[256];
    float s = 0.0f;
    for (int i = threadIdx.x; i < n; i += 256) s += partials[i];
    sm[threadIdx.x] = s;
    __syncthreads();
    for (int off = 128; off > 0; off >>= 1) {
        if ((int)threadIdx.x < off) sm[threadIdx.x] += sm[threadIdx.x + off];
        __syncthreads();
    }
    if (threadIdx.x == 0) out[0] = sm[0];
}

extern "C" void kernel_launch(void* const* d_in, const int* in_sizes, int n_in,
                              void* d_out, int out_size, void* d_ws, size_t ws_size,
                              hipStream_t stream) {
    const float*     scores  = (const float*)d_in[0];
    const long long* werRank = (const long long*)d_in[1];
    // d_in[2] is N (==64); the kernel structure assumes N==64.

    const int B = in_sizes[1] / 64;          // 16384
    const int blocks = 256;                  // 256 blocks x 8 waves = 2048 waves
    const int nWaves = blocks * (256 / 32);  // 2048
    const int bPerWave = B / nWaves;         // 8 (B divisible by 2048)

    float* partials = (float*)d_ws;          // 2048 floats, all written below

    selfMarginLoss_wmma_kernel<<<blocks, 256, 0, stream>>>(
        scores, werRank, partials, B, bPerWave);
    reduce_partials_kernel<<<1, 256, 0, stream>>>(
        partials, (float*)d_out, nWaves);
}
